// Model_Recursive_LSTM_v2_70368744178482
// MI455X (gfx1250) — matrix operand order repaired and checked by reading, not compiled, not run on hardware
//
#include <hip/hip_runtime.h>
#include <hip/hip_bf16.h>

typedef __attribute__((ext_vector_type(8)))  _Float16 v8h;
typedef __attribute__((ext_vector_type(16))) _Float16 v16h;
typedef __attribute__((ext_vector_type(8)))  float    v8f;

__device__ __forceinline__ float eluf(float x) {
    return x > 0.0f ? x : (__expf(x) - 1.0f);
}
__device__ __forceinline__ float sigf(float x) {
    return 1.0f / (1.0f + __expf(-x));
}

// ---------------------------------------------------------------------------
// WMMA GEMM, f16 in / f16 out, f32 accumulate:
//   out[M,N] = act( A1[M,K1] @ W1[N,K1]^T + A2[M,K2] @ W2[N,K2]^T + b1 + b2 )
// K1/K2 are PADDED to multiples of 32; W padding is zero-filled so A padding
// may hold garbage. A1 row i -> (i/subA1)*soA1 + (i%subA1)*siA1 (halves).
// Each wave owns a 16x64 tile; inner loop is pure b128 loads + 4 wmma.
// ---------------------------------------------------------------------------
__global__ void wmma_gemm_h(
    const _Float16* __restrict__ A1, int subA1, long long soA1, long long siA1,
    int K1, const _Float16* __restrict__ W1,
    const _Float16* __restrict__ A2, long long ldA2,
    int K2, const _Float16* __restrict__ W2,
    const float* __restrict__ bias1, const float* __restrict__ bias2,
    _Float16* __restrict__ out, long long ldo, int M, int N, int act)
{
    const int lane = threadIdx.x & 31;
    const int wave = threadIdx.x >> 5;
    const int row0 = (blockIdx.x * 4 + wave) * 16;
    const int n0   = blockIdx.y * 64;
    if (row0 >= M) return;                 // wave-uniform exit, EXEC stays full

    const int ml = lane & 15;
    const int q  = lane >> 4;

    int arow = row0 + ml;
    if (arow >= M) arow = M - 1;           // clamp: reads valid, masked at store
    const long long aoff1 = (long long)(arow / subA1) * soA1 +
                            (long long)(arow % subA1) * siA1;
    const long long aoff2 = (long long)arow * ldA2;

    int wcol[4];
#pragma unroll
    for (int t = 0; t < 4; ++t) {
        const int c = n0 + t * 16 + ml;
        wcol[t] = (c < N) ? c : (N - 1);   // clamp: reads valid, masked at store
    }

    v8f acc[4] = { v8f{}, v8f{}, v8f{}, v8f{} };

    for (int seg = 0; seg < 2; ++seg) {
        const _Float16* __restrict__ A = seg ? A2 : A1;
        const _Float16* __restrict__ W = seg ? W2 : W1;
        const int K = seg ? K2 : K1;
        if (K <= 0 || A == nullptr) continue;

        const _Float16* abase = A + (seg ? aoff2 : aoff1) + q * 8;

        for (int k0 = 0; k0 < K; k0 += 32) {
            // A fragment: halves k0+q*8+[0..7] and k0+16+q*8+[0..7]
            const v8h a0 = *(const v8h*)(abase + k0);
            const v8h a1 = *(const v8h*)(abase + k0 + 16);
            const v16h a = __builtin_shufflevector(a0, a1,
                0, 1, 2, 3, 4, 5, 6, 7, 8, 9, 10, 11, 12, 13, 14, 15);
#pragma unroll
            for (int t = 0; t < 4; ++t) {
                const _Float16* wb = W + (long long)wcol[t] * K + q * 8 + k0;
                const v8h b0 = *(const v8h*)(wb);
                const v8h b1 = *(const v8h*)(wb + 16);
                const v16h b = __builtin_shufflevector(b0, b1,
                    0, 1, 2, 3, 4, 5, 6, 7, 8, 9, 10, 11, 12, 13, 14, 15);
                acc[t] = __builtin_amdgcn_wmma_f32_16x16x32_f16(
                    false, a, false, b, (short)0, acc[t], false, false);
            }
        }
    }

    // C/D layout: lanes 0-15: M=r, lanes 16-31: M=r+8; N=ml
#pragma unroll
    for (int t = 0; t < 4; ++t) {
        const int col = n0 + t * 16 + ml;
        if (col >= N) continue;
        const float bs = (bias1 ? bias1[col] : 0.0f) + (bias2 ? bias2[col] : 0.0f);
#pragma unroll
        for (int r = 0; r < 8; ++r) {
            const int rr = row0 + q * 8 + r;
            if (rr >= M) continue;
            float v = acc[t][r] + bs;
            if (act) v = eluf(v);
            out[(long long)rr * ldo + col] = (_Float16)v;
        }
    }
}

// ---------------------------------------------------------------------------
// weight conversion: f32 (N x K) -> f16 (N x Kp), zero padded
// ---------------------------------------------------------------------------
__global__ void conv_w(const float* __restrict__ W, _Float16* __restrict__ out,
                       int N, int K, int Kp)
{
    const long long idx = (long long)blockIdx.x * blockDim.x + threadIdx.x;
    if (idx >= (long long)N * Kp) return;
    const int n = (int)(idx / Kp);
    const int k = (int)(idx % Kp);
    out[idx] = (k < K) ? (_Float16)W[(long long)n * K + k] : (_Float16)0.0f;
}

// LSTM pointwise: gates f16 (M x 4H, order i,f,g,o), c f32 (M x H),
// h f16 (M x Hp, padded; pads pre-zeroed)
__global__ void lstm_pointwise(const _Float16* __restrict__ gates,
                               float* __restrict__ c, _Float16* __restrict__ h,
                               int M, int H, int Hp)
{
    const long long idx = (long long)blockIdx.x * blockDim.x + threadIdx.x;
    if (idx >= (long long)M * H) return;
    const int r = (int)(idx / H);
    const int k = (int)(idx % H);
    const long long base = (long long)r * 4 * H;
    const float gi = (float)gates[base + k];
    const float gf = (float)gates[base + H + k];
    const float gg = (float)gates[base + 2 * H + k];
    const float go = (float)gates[base + 3 * H + k];
    const float cn = sigf(gf) * c[idx] + sigf(gi) * tanhf(gg);
    c[idx] = cn;
    h[(long long)r * Hp + k] = (_Float16)(sigf(go) * tanhf(cn));
}

__global__ void fill_zero_h(_Float16* __restrict__ p, long long n)
{
    const long long idx = (long long)blockIdx.x * blockDim.x + threadIdx.x;
    if (idx < n) p[idx] = (_Float16)0.0f;
}
__global__ void fill_zero_f(float* __restrict__ p, long long n)
{
    const long long idx = (long long)blockIdx.x * blockDim.x + threadIdx.x;
    if (idx < n) p[idx] = 0.0f;
}

// vec_in (R*5 x 64 f16): row (r,m) -> comps[r, 75 + m*36 + k], zero pad k>=36
__global__ void build_vec_in(const float* __restrict__ comps,
                             _Float16* __restrict__ out, long long n)
{
    const long long idx = (long long)blockIdx.x * blockDim.x + threadIdx.x;
    if (idx >= n) return;
    const long long row = idx / 64;
    const int       k   = (int)(idx % 64);
    const long long r   = row / 5;
    const int       m   = (int)(row % 5);
    out[idx] = (k < 36) ? (_Float16)comps[r * 956 + 75 + m * 36 + k]
                        : (_Float16)0.0f;
}

// x976 (R x 992 f16) = [comps cols 0..75 | h_f | h_b | comps cols 255..956 | 0pad]
__global__ void build_x976(const float* __restrict__ comps,
                           const _Float16* __restrict__ hf,
                           const _Float16* __restrict__ hb,
                           _Float16* __restrict__ out, long long n)
{
    const long long idx = (long long)blockIdx.x * blockDim.x + threadIdx.x;
    if (idx >= n) return;
    const long long r = idx / 992;
    const int       c = (int)(idx % 992);
    _Float16 v;
    if (c < 75)       v = (_Float16)comps[r * 956 + c];
    else if (c < 175) v = hf[r * 128 + (c - 75)];
    else if (c < 275) v = hb[r * 128 + (c - 175)];
    else if (c < 976) v = (_Float16)comps[r * 956 + (c - 20)];  // 255+(c-275)
    else              v = (_Float16)0.0f;
    out[idx] = v;
}

// concat input (M x 384 f16) = [nodes(180) | comps(180) | loop(20) | 0pad]
// matrices have row stride 192 (padded); bcast sources are f32 length-180
__global__ void build_concat(_Float16* __restrict__ out,
                             const _Float16* __restrict__ nodes_mat,
                             const float* __restrict__ nodes_bcast,
                             const _Float16* __restrict__ comps_mat,
                             const float* __restrict__ comps_bcast,
                             const float* __restrict__ loops,
                             int loopBase, int nodesPerB, int M)
{
    const long long idx = (long long)blockIdx.x * blockDim.x + threadIdx.x;
    if (idx >= (long long)M * 384) return;
    const int i = (int)(idx / 384);
    const int c = (int)(idx % 384);
    const int b = i / nodesPerB;
    const int j = i % nodesPerB;
    _Float16 v;
    if (c < 180) {
        v = nodes_bcast ? (_Float16)nodes_bcast[c]
                        : nodes_mat[(long long)i * 192 + c];
    } else if (c < 360) {
        const int k = c - 180;
        v = comps_bcast ? (_Float16)comps_bcast[k]
                        : comps_mat[(long long)i * 192 + k];
    } else if (c < 380) {
        v = (_Float16)loops[(long long)b * 320 + (loopBase + j) * 20 + (c - 360)];
    } else {
        v = (_Float16)0.0f;
    }
    out[idx] = v;
}

// final predict: out[b] = elu( dot(x[b,0:180] (f16, ld 192), Wp) + bp )
__global__ void predict_elu(const _Float16* __restrict__ x,
                            const float* __restrict__ Wp,
                            const float* __restrict__ bp,
                            float* __restrict__ out, int M)
{
    const int b = blockIdx.x * blockDim.x + threadIdx.x;
    if (b >= M) return;
    float s = bp[0];
    const _Float16* row = x + (long long)b * 192;
#pragma unroll 4
    for (int k = 0; k < 180; ++k) s += (float)row[k] * Wp[k];
    out[b] = eluf(s);
}

// ---------------------------------------------------------------------------
extern "C" void kernel_launch(void* const* d_in, const int* in_sizes, int n_in,
                              void* d_out, int out_size, void* d_ws, size_t ws_size,
                              hipStream_t stream)
{
    (void)in_sizes; (void)n_in; (void)out_size; (void)ws_size;

    const int R = 32768;   // B*C = 1024*32

    // ---- inputs (jax-pytree flatten order of setup_inputs dict) ----
    const float* comps = (const float*)d_in[0];
    const float* loops = (const float*)d_in[1];
    const float* Wenc = (const float*)d_in[2];  const float* benc = (const float*)d_in[3];
    const float* Wif  = (const float*)d_in[4];  const float* Whf  = (const float*)d_in[5];
    const float* bif  = (const float*)d_in[6];  const float* bhf  = (const float*)d_in[7];
    const float* Wib  = (const float*)d_in[8];  const float* Whb  = (const float*)d_in[9];
    const float* bib  = (const float*)d_in[10]; const float* bhb  = (const float*)d_in[11];
    const float* We0 = (const float*)d_in[12];  const float* be0 = (const float*)d_in[13];
    const float* We1 = (const float*)d_in[14];  const float* be1 = (const float*)d_in[15];
    const float* We2 = (const float*)d_in[16];  const float* be2 = (const float*)d_in[17];
    const float* We3 = (const float*)d_in[18];  const float* be3 = (const float*)d_in[19];
    const float* Wcx = (const float*)d_in[20];  const float* Wch = (const float*)d_in[21];
    const float* bcx = (const float*)d_in[22];  const float* bch = (const float*)d_in[23];
    const float* Wnx = (const float*)d_in[24];  const float* Wnh = (const float*)d_in[25];
    const float* bnx = (const float*)d_in[26];  const float* bnh = (const float*)d_in[27];
    const float* Wc0 = (const float*)d_in[28];  const float* bc0 = (const float*)d_in[29];
    const float* Wc1 = (const float*)d_in[30];  const float* bc1 = (const float*)d_in[31];
    const float* Wr0 = (const float*)d_in[32];  const float* br0 = (const float*)d_in[33];
    const float* Wr1 = (const float*)d_in[34];  const float* br1 = (const float*)d_in[35];
    const float* Wp  = (const float*)d_in[36];  const float* bp  = (const float*)d_in[37];
    const float* no_comps = (const float*)d_in[38];
    const float* no_nodes = (const float*)d_in[39];

    // ---- workspace carving (units: f16 halves; all offsets multiple of 8) ---
    _Float16* wsh = (_Float16*)d_ws;
    size_t off = 0;
    auto allocH = [&](size_t n) { _Float16* p = wsh + off; off += n; return p; };

    _Float16* vec_in   = allocH(10485760);   // (R*5) x 64
    _Float16* vec_enc  = allocH(10485760);   // (R*5) x 64
    _Float16* h_f      = allocH(4194304);    // R x 128 (pads zeroed)
    _Float16* h_b      = allocH(4194304);    // R x 128
    _Float16* gatesB   = allocH(13107200);   // R x 400
    _Float16* F        = allocH(32505856);   // big alias region (R x 992)
    _Float16* G        = allocH(19922944);   // R x 608 region
    float*    cbuf     = (float*)allocH(6553600);  // R*100 f32 (reused for tree c)

    // aliases within F / G
    _Float16* x976 = F;
    _Float16* e600 = G;                      // R x 608
    _Float16* e350 = F;                      // R x 352 (x976 dead)
    _Float16* e200 = F + 11534336;           // R x 224
    _Float16* ce   = G;                      // R x 192 (e600 dead)
    // tree phase (F region free after ce written)
    _Float16* leaf_h_l = F;                  // 8192 x 192
    _Float16* leaf_gts = F + 1572864;        // 8192 x 720
    _Float16* cat_in   = F + 7471104;        // max 8192 x 384
    _Float16* cl200    = F + 10616832;       // max 8192 x 224
    _Float16* leaf_h   = F + 12451840;       // (1024,8,192)
    _Float16* node_h_l = F + 14024704;       // 4096 x 192
    _Float16* node_gts = F + 14811136;       // 4096 x 720
    _Float16* branch_h = F + 17760256;       // (1024,4,192)
    _Float16* root_h   = F + 18546688;       // 1024 x 192
    _Float16* reg200   = F + 18743296;       // 1024 x 224
    _Float16* reg180   = F + 18972672;       // 1024 x 192

    // f16 weight buffers (zero K-padding)
    _Float16* Wenc_h = allocH(2304);      // 36 x 64
    _Float16* Wif_h  = allocH(25600);     // 400 x 64
    _Float16* Whf_h  = allocH(51200);     // 400 x 128
    _Float16* Wib_h  = allocH(25600);
    _Float16* Whb_h  = allocH(51200);
    _Float16* We0_h  = allocH(595200);    // 600 x 992
    _Float16* We1_h  = allocH(212800);    // 350 x 608
    _Float16* We2_h  = allocH(70400);     // 200 x 352
    _Float16* We3_h  = allocH(40320);     // 180 x 224
    _Float16* Wcx_h  = allocH(138240);    // 720 x 192
    _Float16* Wch_h  = allocH(138240);
    _Float16* Wnx_h  = allocH(138240);
    _Float16* Wnh_h  = allocH(138240);
    _Float16* Wc0_h  = allocH(76800);     // 200 x 384
    _Float16* Wc1_h  = allocH(40320);     // 180 x 224
    _Float16* Wr0_h  = allocH(38400);     // 200 x 192
    _Float16* Wr1_h  = allocH(40320);     // 180 x 224

    auto gemm = [&](const _Float16* A1, int sub, long long so, long long si,
                    int K1, const _Float16* W1,
                    const _Float16* A2, long long ldA2, int K2, const _Float16* W2,
                    const float* b1, const float* b2,
                    _Float16* outp, long long ldo, int M, int N, int act) {
        dim3 g((M + 63) / 64, (N + 63) / 64);
        wmma_gemm_h<<<g, 128, 0, stream>>>(A1, sub, so, si, K1, W1, A2, ldA2, K2, W2,
                                           b1, b2, outp, ldo, M, N, act);
    };
    auto cw = [&](const float* W, _Float16* dst, int N, int K, int Kp) {
        long long n = (long long)N * Kp;
        conv_w<<<(unsigned)((n + 255) / 256), 256, 0, stream>>>(W, dst, N, K, Kp);
    };
    auto zh = [&](_Float16* p, long long n) {
        fill_zero_h<<<(unsigned)((n + 255) / 256), 256, 0, stream>>>(p, n);
    };
    auto zf = [&](float* p, long long n) {
        fill_zero_f<<<(unsigned)((n + 255) / 256), 256, 0, stream>>>(p, n);
    };
    auto lstm_pw = [&](_Float16* g, float* c, _Float16* h, int M, int H, int Hp) {
        long long n = (long long)M * H;
        lstm_pointwise<<<(unsigned)((n + 255) / 256), 256, 0, stream>>>(g, c, h, M, H, Hp);
    };

    // ===== Phase 0: weight conversion (tiny, L2-resident afterwards) ========
    cw(Wenc, Wenc_h, 36, 36, 64);
    cw(Wif, Wif_h, 400, 36, 64);   cw(Whf, Whf_h, 400, 100, 128);
    cw(Wib, Wib_h, 400, 36, 64);   cw(Whb, Whb_h, 400, 100, 128);
    cw(We0, We0_h, 600, 976, 992); cw(We1, We1_h, 350, 600, 608);
    cw(We2, We2_h, 200, 350, 352); cw(We3, We3_h, 180, 200, 224);
    cw(Wcx, Wcx_h, 720, 180, 192); cw(Wch, Wch_h, 720, 180, 192);
    cw(Wnx, Wnx_h, 720, 180, 192); cw(Wnh, Wnh_h, 720, 180, 192);
    cw(Wc0, Wc0_h, 200, 380, 384); cw(Wc1, Wc1_h, 180, 200, 224);
    cw(Wr0, Wr0_h, 200, 180, 192); cw(Wr1, Wr1_h, 180, 200, 224);

    // ===== Phase 1: encoder linear on the 5 "vector" slices =================
    {
        long long n = (long long)R * 5 * 64;
        build_vec_in<<<(unsigned)((n + 255) / 256), 256, 0, stream>>>(comps, vec_in, n);
    }
    gemm(vec_in, 1, 64, 0, 64, Wenc_h, nullptr, 0, 0, nullptr,
         benc, nullptr, vec_enc, 64, R * 5, 36, 0);

    // ===== Phase 2: BiLSTM (H=100, seq=5) ===================================
    zh(h_f, (long long)R * 128); zf(cbuf, (long long)R * 100);
    for (int t = 0; t < 5; ++t) {
        gemm(vec_enc + t * 64, 1, 320, 0, 64, Wif_h, h_f, 128, 128, Whf_h,
             bif, bhf, gatesB, 400, R, 400, 0);
        lstm_pw(gatesB, cbuf, h_f, R, 100, 128);
    }
    zh(h_b, (long long)R * 128); zf(cbuf, (long long)R * 100);
    for (int t = 0; t < 5; ++t) {
        gemm(vec_enc + (4 - t) * 64, 1, 320, 0, 64, Wib_h, h_b, 128, 128, Whb_h,
             bib, bhb, gatesB, 400, R, 400, 0);
        lstm_pw(gatesB, cbuf, h_b, R, 100, 128);
    }

    // ===== Phase 3: assemble 992-wide (976 + pad) features ==================
    {
        long long n = (long long)R * 992;
        build_x976<<<(unsigned)((n + 255) / 256), 256, 0, stream>>>(comps, h_f, h_b,
                                                                    x976, n);
    }

    // ===== Phase 4: embedding MLP 976->600->350->200->180 (ELU) =============
    gemm(x976, 1, 992, 0, 992, We0_h, nullptr, 0, 0, nullptr, be0, nullptr, e600, 608, R, 600, 1);
    gemm(e600, 1, 608, 0, 608, We1_h, nullptr, 0, 0, nullptr, be1, nullptr, e350, 352, R, 350, 1);
    gemm(e350, 1, 352, 0, 352, We2_h, nullptr, 0, 0, nullptr, be2, nullptr, e200, 224, R, 200, 1);
    gemm(e200, 1, 224, 0, 224, We3_h, nullptr, 0, 0, nullptr, be3, nullptr, ce,   192, R, 180, 1);

    // ===== Phase 5: 8 leaves batched (M=8192), comps_lstm seq=4 =============
    zh(leaf_h_l, 8192LL * 192); zf(cbuf, 8192LL * 180);
    for (int t = 0; t < 4; ++t) {
        // row (b,j): ce[b, 4j+t, :] -> base+t*192, outer 32*192, inner 4*192
        gemm(ce + t * 192, 8, 6144, 768, 192, Wcx_h, leaf_h_l, 192, 192, Wch_h,
             bcx, bch, leaf_gts, 720, 8192, 720, 0);
        lstm_pw(leaf_gts, cbuf, leaf_h_l, 8192, 180, 192);
    }
    {
        long long n = 8192LL * 384;
        build_concat<<<(unsigned)((n + 255) / 256), 256, 0, stream>>>(
            cat_in, nullptr, no_nodes, leaf_h_l, nullptr, loops, 5, 8, 8192);
    }
    gemm(cat_in, 1, 384, 0, 384, Wc0_h, nullptr, 0, 0, nullptr, bc0, nullptr, cl200, 224, 8192, 200, 1);
    gemm(cl200,  1, 224, 0, 224, Wc1_h, nullptr, 0, 0, nullptr, bc1, nullptr, leaf_h, 192, 8192, 180, 1);

    // ===== Phase 6: 4 branches batched (M=4096), nodes_lstm seq=2 ===========
    zh(node_h_l, 4096LL * 192); zf(cbuf, 4096LL * 180);
    for (int t = 0; t < 2; ++t) {
        // row (b,br): leaf_h[b, 2br+t, :] -> base+t*192, outer 8*192, inner 2*192
        gemm(leaf_h + t * 192, 4, 1536, 384, 192, Wnx_h, node_h_l, 192, 192, Wnh_h,
             bnx, bnh, node_gts, 720, 4096, 720, 0);
        lstm_pw(node_gts, cbuf, node_h_l, 4096, 180, 192);
    }
    {
        long long n = 4096LL * 384;
        build_concat<<<(unsigned)((n + 255) / 256), 256, 0, stream>>>(
            cat_in, node_h_l, nullptr, nullptr, no_comps, loops, 1, 4, 4096);
    }
    gemm(cat_in, 1, 384, 0, 384, Wc0_h, nullptr, 0, 0, nullptr, bc0, nullptr, cl200, 224, 4096, 200, 1);
    gemm(cl200,  1, 224, 0, 224, Wc1_h, nullptr, 0, 0, nullptr, bc1, nullptr, branch_h, 192, 4096, 180, 1);

    // ===== Phase 7: root, nodes_lstm over 4 branch states ===================
    zh(node_h_l, 1024LL * 192); zf(cbuf, 1024LL * 180);
    for (int t = 0; t < 4; ++t) {
        gemm(branch_h + t * 192, 1, 768, 0, 192, Wnx_h, node_h_l, 192, 192, Wnh_h,
             bnx, bnh, node_gts, 720, 1024, 720, 0);
        lstm_pw(node_gts, cbuf, node_h_l, 1024, 180, 192);
    }
    {
        long long n = 1024LL * 384;
        build_concat<<<(unsigned)((n + 255) / 256), 256, 0, stream>>>(
            cat_in, node_h_l, nullptr, nullptr, no_comps, loops, 0, 1, 1024);
    }
    gemm(cat_in, 1, 384, 0, 384, Wc0_h, nullptr, 0, 0, nullptr, bc0, nullptr, cl200, 224, 1024, 200, 1);
    gemm(cl200,  1, 224, 0, 224, Wc1_h, nullptr, 0, 0, nullptr, bc1, nullptr, root_h, 192, 1024, 180, 1);

    // ===== Phase 8: regression + predict ====================================
    gemm(root_h, 1, 192, 0, 192, Wr0_h, nullptr, 0, 0, nullptr, br0, nullptr, reg200, 224, 1024, 200, 1);
    gemm(reg200, 1, 224, 0, 224, Wr1_h, nullptr, 0, 0, nullptr, br1, nullptr, reg180, 192, 1024, 180, 1);
    predict_elu<<<4, 256, 0, stream>>>(reg180, Wp, bp, (float*)d_out, 1024);
}
